// transformer_35124242546801
// MI455X (gfx1250) — compile-verified, matrix operand order checked by reading, not run on hardware
//
#include <hip/hip_runtime.h>
#include <math.h>

// ---------------------------------------------------------------------------
// Transformer enc-dec forward for MI455X (gfx1250, wave32, WMMA f16->f32).
// S=1024 B=8 D=512 H=8 K=64 F=2048, 6 shared enc layers + 6 shared dec layers.
// ---------------------------------------------------------------------------

#define SEQ 1024
#define BAT 8
#define DM  512
#define NH  8
#define DK  64
#define DFF 2048
#define TOK (SEQ * BAT)          // 8192 tokens

typedef _Float16 half_t;
typedef __attribute__((ext_vector_type(16))) _Float16 v16h;
typedef __attribute__((ext_vector_type(8)))  float    v8f;
typedef int i4_t __attribute__((vector_size(16)));

#define AS1 __attribute__((address_space(1)))
#define AS3 __attribute__((address_space(3)))

// Optional CDNA5 async global->LDS path (no VGPR transit, ASYNCcnt-tracked).
#if defined(__has_builtin)
#if __has_builtin(__builtin_amdgcn_global_load_async_to_lds_b128) && \
    __has_builtin(__builtin_amdgcn_s_wait_asynccnt)
#define USE_ASYNC_LDS 1
#endif
#endif
#ifndef USE_ASYNC_LDS
#define USE_ASYNC_LDS 0
#endif

// Copy 16 bytes global -> LDS (per lane).
__device__ __forceinline__ void stage16(const half_t* g, half_t* l) {
#if USE_ASYNC_LDS
  __builtin_amdgcn_global_load_async_to_lds_b128(
      (AS1 i4_t*)(AS1 void*)g, (AS3 i4_t*)(AS3 void*)l, 0, 0);
#else
  *reinterpret_cast<uint4*>(l) = *reinterpret_cast<const uint4*>(g);
#endif
}
__device__ __forceinline__ void stage_wait() {
#if USE_ASYNC_LDS
  __builtin_amdgcn_s_wait_asynccnt(0);
#endif
}

// Element index (within a 32-wide K slice) for pair p of a 16-bit A/B WMMA
// fragment: lanes 0-15 hold K 0..7 & 16..23, lanes 16-31 hold K 8..15 & 24..31.
__device__ __forceinline__ int frag_kk(int lane, int p) {
  return ((lane >> 4) << 3) + ((p >> 2) << 4) + ((p & 3) << 1);
}

// Load one v16h fragment from LDS; base points at this lane's row start
// (pairs are contiguous 4-byte words -> ds_load_b32/b128 after coalescing).
__device__ __forceinline__ v16h frag_from_lds(const half_t* base, int lane) {
  v16h f;
  unsigned int* fp = reinterpret_cast<unsigned int*>(&f);
#pragma unroll
  for (int p = 0; p < 8; ++p)
    fp[p] = *reinterpret_cast<const unsigned int*>(base + frag_kk(lane, p));
  return f;
}

// ---------------------------------------------------------------------------
// Generic GEMM: C[M,N] = act(A[M,Kd] * B[Kd,N] + bias), A/B f16, accum f32.
// Block tile 128x128, 8 waves as 2(M) x 4(N), wave tile 64x32 (4x2 WMMA):
// 8 WMMAs per K-step against 6 fragment loads.
// ---------------------------------------------------------------------------
template <bool RELU, bool OUTF16>
__global__ __launch_bounds__(256) void gemm_kernel(
    const half_t* __restrict__ A, const half_t* __restrict__ B,
    const float* __restrict__ bias, void* __restrict__ Cout, int N, int Kd) {
  __shared__ __attribute__((aligned(16))) half_t As[128 * 40];
  __shared__ __attribute__((aligned(16))) half_t Bs[128 * 40];
  const int tid = threadIdx.x, lane = tid & 31, wave = tid >> 5;
  const int wm = wave >> 2, wn = wave & 3;
  const int m0 = blockIdx.x * 128, n0 = blockIdx.y * 128;
  v8f c[4][2] = {};

  for (int k0 = 0; k0 < Kd; k0 += 32) {
    __syncthreads();
    // Stage A tile 128x32 (row-major, stride 40 to dodge bank conflicts).
#pragma unroll
    for (int i = 0; i < 2; ++i) {
      int e = tid + i * 256;
      int row = e >> 2, c8 = (e & 3) << 3;
      stage16(&A[(size_t)(m0 + row) * Kd + k0 + c8], &As[row * 40 + c8]);
    }
    // Stage B tile 32x128 transposed -> Bs[n][k].
#pragma unroll
    for (int i = 0; i < 2; ++i) {
      int e = tid + i * 256;
      int kk = e >> 4, n8 = (e & 15) << 3;
      union { uint4 v; half_t h[8]; } u;
      u.v = *reinterpret_cast<const uint4*>(&B[(size_t)(k0 + kk) * N + n0 + n8]);
#pragma unroll
      for (int j = 0; j < 8; ++j) Bs[(n8 + j) * 40 + kk] = u.h[j];
    }
    if (k0 + 32 < Kd)  // pull next slices toward L2/L0
      __builtin_prefetch(&A[(size_t)(m0 + (tid >> 1)) * Kd + k0 + 32], 0, 0);
    stage_wait();
    __syncthreads();

    v16h af[4], bf[2];
#pragma unroll
    for (int i = 0; i < 4; ++i)
      af[i] = frag_from_lds(&As[(wm * 64 + i * 16 + (lane & 15)) * 40], lane);
#pragma unroll
    for (int j = 0; j < 2; ++j)
      bf[j] = frag_from_lds(&Bs[(wn * 32 + j * 16 + (lane & 15)) * 40], lane);
#pragma unroll
    for (int i = 0; i < 4; ++i)
#pragma unroll
      for (int j = 0; j < 2; ++j)
        c[i][j] = __builtin_amdgcn_wmma_f32_16x16x32_f16(
            false, af[i], false, bf[j], (short)0, c[i][j], false, false);
  }

#pragma unroll
  for (int i = 0; i < 4; ++i)
#pragma unroll
    for (int j = 0; j < 2; ++j)
#pragma unroll
      for (int r = 0; r < 8; ++r) {
        int row = m0 + wm * 64 + i * 16 + r + ((lane >> 4) << 3);
        int col = n0 + wn * 32 + j * 16 + (lane & 15);
        float v = c[i][j][r] + bias[col];
        if (RELU) v = fmaxf(v, 0.0f);
        if (OUTF16)
          reinterpret_cast<half_t*>(Cout)[(size_t)row * N + col] = (half_t)v;
        else
          reinterpret_cast<float*>(Cout)[(size_t)row * N + col] = v;
      }
}

// ---------------------------------------------------------------------------
// Per-head projection: Out[h][b][s][:] = A[t=s*B+b, :512] * W[h][:512][:64].
// Block tile 128x64, 8 waves as 4(M) x 2(N), wave tile 32x32.
// ---------------------------------------------------------------------------
__global__ __launch_bounds__(256) void proj_head_kernel(
    const half_t* __restrict__ A, const half_t* __restrict__ W,
    half_t* __restrict__ Out) {
  __shared__ __attribute__((aligned(16))) half_t As[128 * 40];
  __shared__ __attribute__((aligned(16))) half_t Bs[64 * 40];
  const int tid = threadIdx.x, lane = tid & 31, wave = tid >> 5;
  const int wm = wave >> 1, wn = wave & 1;
  const int m0 = blockIdx.x * 128;
  const half_t* Wh = W + (size_t)blockIdx.y * DM * DK;
  v8f c[2][2] = {};

  for (int k0 = 0; k0 < DM; k0 += 32) {
    __syncthreads();
#pragma unroll
    for (int i = 0; i < 2; ++i) {
      int e = tid + i * 256;
      int row = e >> 2, c8 = (e & 3) << 3;
      stage16(&A[(size_t)(m0 + row) * DM + k0 + c8], &As[row * 40 + c8]);
    }
    {
      int kk = tid >> 3, n8 = (tid & 7) << 3;
      union { uint4 v; half_t h[8]; } u;
      u.v = *reinterpret_cast<const uint4*>(&Wh[(size_t)(k0 + kk) * DK + n8]);
#pragma unroll
      for (int j = 0; j < 8; ++j) Bs[(n8 + j) * 40 + kk] = u.h[j];
    }
    stage_wait();
    __syncthreads();

    v16h af[2], bf[2];
#pragma unroll
    for (int i = 0; i < 2; ++i)
      af[i] = frag_from_lds(&As[(wm * 32 + i * 16 + (lane & 15)) * 40], lane);
#pragma unroll
    for (int j = 0; j < 2; ++j)
      bf[j] = frag_from_lds(&Bs[(wn * 32 + j * 16 + (lane & 15)) * 40], lane);
#pragma unroll
    for (int i = 0; i < 2; ++i)
#pragma unroll
      for (int j = 0; j < 2; ++j)
        c[i][j] = __builtin_amdgcn_wmma_f32_16x16x32_f16(
            false, af[i], false, bf[j], (short)0, c[i][j], false, false);
  }

  const size_t hb = (size_t)blockIdx.y * BAT;
#pragma unroll
  for (int i = 0; i < 2; ++i)
#pragma unroll
    for (int j = 0; j < 2; ++j)
#pragma unroll
      for (int r = 0; r < 8; ++r) {
        int t = m0 + wm * 32 + i * 16 + r + ((lane >> 4) << 3);
        int col = wn * 32 + j * 16 + (lane & 15);
        int s = t >> 3, b = t & 7;
        Out[((hb + b) * SEQ + s) * DK + col] = (half_t)c[i][j][r];
      }
}

// ---------------------------------------------------------------------------
// Flash-style attention. grid = (S/128, H*B); wave owns 16 query rows.
// Online softmax with row stats reduced by shfl_xor across the 16-lane
// C/D row groups; P goes through a per-wave LDS slab to become an A fragment.
// ---------------------------------------------------------------------------
__global__ __launch_bounds__(256) void attention_kernel(
    const half_t* __restrict__ Q, const half_t* __restrict__ Km,
    const half_t* __restrict__ V, float* __restrict__ Out) {
  __shared__ __attribute__((aligned(16))) half_t Ks[64 * 72];
  __shared__ __attribute__((aligned(16))) half_t Vt[64 * 72];
  __shared__ __attribute__((aligned(16))) half_t Ps[8 * 16 * 72];
  const int tid = threadIdx.x, lane = tid & 31, wave = tid >> 5;
  const int hb = blockIdx.y;
  const int h = hb >> 3, b = hb & 7;
  const size_t base = (size_t)hb * SEQ * DK;
  const int q0 = blockIdx.x * 128 + wave * 16;
  const float scale = 0.044194173824159216f;  // quirk: 1/sqrt(D), not 1/sqrt(K)

  // Q fragments (A layout) straight from global: pairs are contiguous.
  v16h qa[2];
  {
    const unsigned int* qp = reinterpret_cast<const unsigned int*>(
        Q + base + (size_t)(q0 + (lane & 15)) * DK);
#pragma unroll
    for (int st = 0; st < 2; ++st) {
      unsigned int* fp = reinterpret_cast<unsigned int*>(&qa[st]);
#pragma unroll
      for (int p = 0; p < 8; ++p) fp[p] = qp[(frag_kk(lane, p) + st * 32) >> 1];
    }
  }

  v8f o[4] = {};
  float mrow[8], lrow[8];
#pragma unroll
  for (int r = 0; r < 8; ++r) { mrow[r] = -1e30f; lrow[r] = 0.0f; }
  half_t* Pw = &Ps[wave * 16 * 72];

  for (int kt = 0; kt < SEQ / 64; ++kt) {
    __syncthreads();
    // Stage K tile [key][kdim] (async path) and V^T tile [kcol][key].
#pragma unroll
    for (int i = 0; i < 2; ++i) {
      int e = tid + i * 256;
      int key = e >> 3, c8 = (e & 7) << 3;
      stage16(&Km[base + (size_t)(kt * 64 + key) * DK + c8], &Ks[key * 72 + c8]);
      union { uint4 v; half_t h8[8]; } u;
      u.v = *reinterpret_cast<const uint4*>(&V[base + (size_t)(kt * 64 + key) * DK + c8]);
#pragma unroll
      for (int j = 0; j < 8; ++j) Vt[(c8 + j) * 72 + key] = u.h8[j];
    }
    stage_wait();
    __syncthreads();

    // Scores 16x64 = Q(16x64) * K^T.
    v8f sc[4] = {};
#pragma unroll
    for (int st = 0; st < 2; ++st)
#pragma unroll
      for (int n = 0; n < 4; ++n) {
        v16h bf = frag_from_lds(&Ks[(n * 16 + (lane & 15)) * 72 + st * 32], lane);
        sc[n] = __builtin_amdgcn_wmma_f32_16x16x32_f16(
            false, qa[st], false, bf, (short)0, sc[n], false, false);
      }
#pragma unroll
    for (int n = 0; n < 4; ++n) sc[n] *= scale;

    // Online softmax per row (slot r lives in a 16-lane group).
#pragma unroll
    for (int r = 0; r < 8; ++r) {
      float tmax = fmaxf(fmaxf(sc[0][r], sc[1][r]), fmaxf(sc[2][r], sc[3][r]));
#pragma unroll
      for (int m = 8; m >= 1; m >>= 1) tmax = fmaxf(tmax, __shfl_xor(tmax, m, 32));
      float mnew = fmaxf(mrow[r], tmax);
      float corr = __expf(mrow[r] - mnew);
      mrow[r] = mnew;
      lrow[r] *= corr;
#pragma unroll
      for (int n = 0; n < 4; ++n) o[n][r] *= corr;
      float rs = 0.0f;
      int prow = r + ((lane >> 4) << 3);
#pragma unroll
      for (int n = 0; n < 4; ++n) {
        float pv = __expf(sc[n][r] - mnew);
        rs += pv;
        Pw[prow * 72 + n * 16 + (lane & 15)] = (half_t)pv;
      }
#pragma unroll
      for (int m = 8; m >= 1; m >>= 1) rs += __shfl_xor(rs, m, 32);
      lrow[r] += rs;
    }

    // O += P(16x64) * V(64x64). Same-wave DS RAW is in-order on CDNA5.
#pragma unroll
    for (int st = 0; st < 2; ++st) {
      v16h pa = frag_from_lds(&Pw[(lane & 15) * 72 + st * 32], lane);
#pragma unroll
      for (int n = 0; n < 4; ++n) {
        v16h vb = frag_from_lds(&Vt[(n * 16 + (lane & 15)) * 72 + st * 32], lane);
        o[n] = __builtin_amdgcn_wmma_f32_16x16x32_f16(
            false, pa, false, vb, (short)0, o[n], false, false);
      }
    }
  }

  // Normalize and scatter to [s][b][h*K+k] f32 (heads concat directly).
#pragma unroll
  for (int n = 0; n < 4; ++n)
#pragma unroll
    for (int r = 0; r < 8; ++r) {
      int s = q0 + r + ((lane >> 4) << 3);
      int k = n * 16 + (lane & 15);
      Out[((size_t)s * BAT + b) * DM + h * DK + k] = o[n][r] / lrow[r];
    }
}

// ---------------------------------------------------------------------------
// Fused residual + LayerNorm. One block per token; writes f32 + f16 copies.
// ---------------------------------------------------------------------------
__global__ __launch_bounds__(256) void add_ln_kernel(
    const float* __restrict__ X, const float* __restrict__ Yd,
    const float* __restrict__ g, const float* __restrict__ bta,
    float* __restrict__ Of, half_t* __restrict__ Oh) {
  __shared__ float ss[8], qq[8];
  const int i = threadIdx.x;
  const size_t off = (size_t)blockIdx.x * DM;
  float a0 = X[off + i] + Yd[off + i];
  float a1 = X[off + i + 256] + Yd[off + i + 256];
  float s = a0 + a1, q = a0 * a0 + a1 * a1;
#pragma unroll
  for (int m = 16; m >= 1; m >>= 1) {
    s += __shfl_xor(s, m, 32);
    q += __shfl_xor(q, m, 32);
  }
  if ((i & 31) == 0) { ss[i >> 5] = s; qq[i >> 5] = q; }
  __syncthreads();
  float S = 0.0f, Q2 = 0.0f;
#pragma unroll
  for (int w = 0; w < 8; ++w) { S += ss[w]; Q2 += qq[w]; }
  float mean = S * (1.0f / DM);
  float var = Q2 * (1.0f / DM) - mean * mean;
  float inv = rsqrtf(var + 1e-5f);
  float o0 = (a0 - mean) * inv * g[i] + bta[i];
  float o1 = (a1 - mean) * inv * g[i + 256] + bta[i + 256];
  Of[off + i] = o0;
  Of[off + i + 256] = o1;
  Oh[off + i] = (half_t)o0;
  Oh[off + i + 256] = (half_t)o1;
}

__global__ void f32_to_f16_kernel(const float* __restrict__ s,
                                  half_t* __restrict__ d, int n) {
  int i = blockIdx.x * blockDim.x + threadIdx.x;
  if (i < n) d[i] = (half_t)s[i];
}

__global__ void copy_f16_kernel(const half_t* __restrict__ s,
                                half_t* __restrict__ d, int n) {
  int i = blockIdx.x * blockDim.x + threadIdx.x;
  if (i < n) d[i] = s[i];
}

// ---------------------------------------------------------------------------
// Host orchestration.
// ---------------------------------------------------------------------------
extern "C" void kernel_launch(void* const* d_in, const int* in_sizes, int n_in,
                              void* d_out, int out_size, void* d_ws,
                              size_t ws_size, hipStream_t stream) {
  (void)in_sizes; (void)n_in; (void)out_size; (void)ws_size;

  // d_in order: x, enc{sa_Wq,sa_Wk,sa_Wv,ff_W1,ff_b1,ff_W2,ff_b2,
  //                   ln1_g,ln1_b,ln2_g,ln2_b,ln3_g,ln3_b},
  //             dec{sa_Wq,sa_Wk,sa_Wv,ca_Wq,ca_Wk,ca_Wv,ff_W1,ff_b1,ff_W2,
  //                 ff_b2,ln1_g,ln1_b,ln2_g,ln2_b,ln3_g,ln3_b}
  const float* x_in = (const float*)d_in[0];
  const float* eWq = (const float*)d_in[1];
  const float* eWk = (const float*)d_in[2];
  const float* eWv = (const float*)d_in[3];
  const float* eW1 = (const float*)d_in[4];
  const float* eb1 = (const float*)d_in[5];
  const float* eW2 = (const float*)d_in[6];
  const float* eb2 = (const float*)d_in[7];
  const float* eg1 = (const float*)d_in[8];
  const float* ebe1 = (const float*)d_in[9];
  const float* eg2 = (const float*)d_in[10];
  const float* ebe2 = (const float*)d_in[11];
  const float* dWq = (const float*)d_in[14];
  const float* dWk = (const float*)d_in[15];
  const float* dWv = (const float*)d_in[16];
  const float* cWq = (const float*)d_in[17];
  const float* cWk = (const float*)d_in[18];
  const float* cWv = (const float*)d_in[19];
  const float* dW1 = (const float*)d_in[20];
  const float* db1 = (const float*)d_in[21];
  const float* dW2 = (const float*)d_in[22];
  const float* db2 = (const float*)d_in[23];
  const float* dg1 = (const float*)d_in[24];
  const float* dbe1 = (const float*)d_in[25];
  const float* dg2 = (const float*)d_in[26];
  const float* dbe2 = (const float*)d_in[27];
  const float* dg3 = (const float*)d_in[28];
  const float* dbe3 = (const float*)d_in[29];

  const int HDK = NH * DM * DK;  // 262144
  const int DFE = DM * DFF;      // 1048576
  const int TD = TOK * DM;       // 4194304
  const int TF = TOK * DFF;      // 16777216

  char* p = (char*)d_ws;
  auto carve = [&](size_t bytes) {
    void* r = (void*)p;
    p += (bytes + 255) & ~(size_t)255;
    return r;
  };
  half_t* wq_e = (half_t*)carve((size_t)HDK * 2);
  half_t* wk_e = (half_t*)carve((size_t)HDK * 2);
  half_t* wv_e = (half_t*)carve((size_t)HDK * 2);
  half_t* w1_e = (half_t*)carve((size_t)DFE * 2);
  half_t* w2_e = (half_t*)carve((size_t)DFE * 2);
  half_t* wq_d = (half_t*)carve((size_t)HDK * 2);
  half_t* wk_d = (half_t*)carve((size_t)HDK * 2);
  half_t* wv_d = (half_t*)carve((size_t)HDK * 2);
  half_t* cwq_d = (half_t*)carve((size_t)HDK * 2);
  half_t* cwk_d = (half_t*)carve((size_t)HDK * 2);
  half_t* cwv_d = (half_t*)carve((size_t)HDK * 2);
  half_t* w1_d = (half_t*)carve((size_t)DFE * 2);
  half_t* w2_d = (half_t*)carve((size_t)DFE * 2);
  half_t* xh   = (half_t*)carve((size_t)TD * 2);
  half_t* memh = (half_t*)carve((size_t)TD * 2);
  half_t* Qb   = (half_t*)carve((size_t)TD * 2);
  half_t* Kb   = (half_t*)carve((size_t)TD * 2);
  half_t* Vb   = (half_t*)carve((size_t)TD * 2);
  half_t* hh   = (half_t*)carve((size_t)TF * 2);
  float* x0   = (float*)carve((size_t)TD * 4);
  float* ybuf = (float*)carve((size_t)TD * 4);
  float* zbuf = (float*)carve((size_t)TD * 4);
  float* attn = (float*)carve((size_t)TD * 4);

  auto cvt = [&](const float* s, half_t* d, int n) {
    f32_to_f16_kernel<<<(n + 255) / 256, 256, 0, stream>>>(s, d, n);
  };
  cvt(eWq, wq_e, HDK); cvt(eWk, wk_e, HDK); cvt(eWv, wv_e, HDK);
  cvt(eW1, w1_e, DFE); cvt(eW2, w2_e, DFE);
  cvt(dWq, wq_d, HDK); cvt(dWk, wk_d, HDK); cvt(dWv, wv_d, HDK);
  cvt(cWq, cwq_d, HDK); cvt(cWk, cwk_d, HDK); cvt(cWv, cwv_d, HDK);
  cvt(dW1, w1_d, DFE); cvt(dW2, w2_d, DFE);
  cvt(x_in, xh, TD);

  const dim3 projGrid(TOK / 128, NH);
  const dim3 attnGrid(SEQ / 128, NH * BAT);
  const dim3 ffn1Grid(TOK / 128, DFF / 128);
  const dim3 ffn2Grid(TOK / 128, DM / 128);

  const float* xf = x_in;
  for (int l = 0; l < 6; ++l) {  // encoder (shared params)
    proj_head_kernel<<<projGrid, 256, 0, stream>>>(xh, wq_e, Qb);
    proj_head_kernel<<<projGrid, 256, 0, stream>>>(xh, wk_e, Kb);
    proj_head_kernel<<<projGrid, 256, 0, stream>>>(xh, wv_e, Vb);
    attention_kernel<<<attnGrid, 256, 0, stream>>>(Qb, Kb, Vb, attn);
    add_ln_kernel<<<TOK, 256, 0, stream>>>(xf, attn, eg1, ebe1, ybuf, xh);
    gemm_kernel<true, true><<<ffn1Grid, 256, 0, stream>>>(xh, w1_e, eb1, hh, DFF, DM);
    gemm_kernel<false, false><<<ffn2Grid, 256, 0, stream>>>(hh, w2_e, eb2, attn, DM, DFF);
    add_ln_kernel<<<TOK, 256, 0, stream>>>(ybuf, attn, eg2, ebe2, x0, xh);
    xf = x0;
  }
  copy_f16_kernel<<<(TD + 255) / 256, 256, 0, stream>>>(xh, memh, TD);

  for (int l = 0; l < 6; ++l) {  // decoder (shared params)
    proj_head_kernel<<<projGrid, 256, 0, stream>>>(xh, wq_d, Qb);
    proj_head_kernel<<<projGrid, 256, 0, stream>>>(xh, wk_d, Kb);
    proj_head_kernel<<<projGrid, 256, 0, stream>>>(xh, wv_d, Vb);
    attention_kernel<<<attnGrid, 256, 0, stream>>>(Qb, Kb, Vb, attn);
    add_ln_kernel<<<TOK, 256, 0, stream>>>(xf, attn, dg1, dbe1, ybuf, xh);

    proj_head_kernel<<<projGrid, 256, 0, stream>>>(xh, cwq_d, Qb);
    proj_head_kernel<<<projGrid, 256, 0, stream>>>(memh, cwk_d, Kb);
    proj_head_kernel<<<projGrid, 256, 0, stream>>>(memh, cwv_d, Vb);
    attention_kernel<<<attnGrid, 256, 0, stream>>>(Qb, Kb, Vb, attn);
    add_ln_kernel<<<TOK, 256, 0, stream>>>(ybuf, attn, dg2, dbe2, zbuf, xh);

    gemm_kernel<true, true><<<ffn1Grid, 256, 0, stream>>>(xh, w1_d, db1, hh, DFF, DM);
    gemm_kernel<false, false><<<ffn2Grid, 256, 0, stream>>>(hh, w2_d, db2, attn, DM, DFF);
    float* outb = (l == 5) ? (float*)d_out : x0;
    add_ln_kernel<<<TOK, 256, 0, stream>>>(zbuf, attn, dg3, dbe3, outb, xh);
    xf = x0;
  }
}
